// Head_16200616641032
// MI455X (gfx1250) — matrix operand order repaired and checked by reading, not compiled, not run on hardware
//
#include <hip/hip_runtime.h>
#include <math.h>

// Problem constants (match reference)
#define HS   64
#define CC   1024
#define BB   8
#define TT   2048

typedef __attribute__((ext_vector_type(16))) _Float16 v16h;
typedef __attribute__((ext_vector_type(8)))  _Float16 v8h;
typedef __attribute__((ext_vector_type(2)))  __fp16   fp16x2;  // cvt_pkrtz result type
typedef __attribute__((ext_vector_type(8)))  float    v8f;

union AFrag {                 // A-operand: assemble from packed converts / 16B loads
  v16h   v;
  fp16x2 p[8];
  struct { v8h lo, hi; } s;
};

// ---------------------------------------------------------------------------
// Transpose + convert weights: W[C][H] f32  ->  Wt[H][C] f16
// ---------------------------------------------------------------------------
__global__ void cvt_w_kernel(const float* __restrict__ Wq,
                             const float* __restrict__ Wk,
                             const float* __restrict__ Wv,
                             _Float16* __restrict__ wqt,
                             _Float16* __restrict__ wkt,
                             _Float16* __restrict__ wvt) {
  int i = blockIdx.x * blockDim.x + threadIdx.x;     // over HS*CC
  if (i >= HS * CC) return;
  int h = i / CC;
  int c = i % CC;
  wqt[h * CC + c] = (_Float16)Wq[c * HS + h];
  wkt[h * CC + c] = (_Float16)Wk[c * HS + h];
  wvt[h * CC + c] = (_Float16)Wv[c * HS + h];
}

// ---------------------------------------------------------------------------
// QKV projection: one wave32 computes a 16-row tile of q,k,v (each 16x64).
//   A (16x32 f16) built in-register from f32 x via packed converts
//   (x is read from HBM exactly once -> this kernel rides the 23.3 TB/s roof).
//   B (32x16 f16) loaded contiguously from transposed weights (L2-resident).
// Outputs: qh,kh as f16 [B][T][HS]; v stored transposed vt [B][HS][T].
// ---------------------------------------------------------------------------
__global__ __launch_bounds__(32) void qkv_wmma_kernel(
    const float*    __restrict__ x,    // [B][T][C] f32
    const _Float16* __restrict__ wqt,  // [HS][C]
    const _Float16* __restrict__ wkt,
    const _Float16* __restrict__ wvt,
    _Float16* __restrict__ qh,         // [B][T][HS]
    _Float16* __restrict__ kh,         // [B][T][HS]
    _Float16* __restrict__ vt)         // [B][HS][T]
{
  const int lane = threadIdx.x;
  const int tile = blockIdx.x % (TT / 16);
  const int b    = blockIdx.x / (TT / 16);
  const int r0   = tile * 16;

  const int m    = lane & 15;
  const int hiL  = lane >> 4;          // 0: lanes 0-15, 1: lanes 16-31
  const int aoff = hiL ? 16 : 0;       // A-frag K offset: contiguous k = 16*hiL + s
  const int boff = hiL ? 16 : 0;       // B-frag K offset: contiguous k = 16*hiL + s

  v8f accq[4] = {}; v8f acck[4] = {}; v8f accv[4] = {};

  const float* xrow = x + (size_t)(b * TT + r0 + m) * CC;

  for (int k0 = 0; k0 < CC; k0 += 32) {
    // A fragment: 16 f32 -> f16 via packed converts (v_cvt_pk_rtz_f16_f32)
    v8f xlo = *(const v8f*)(xrow + k0 + aoff);
    v8f xhi = *(const v8f*)(xrow + k0 + aoff + 8);
    AFrag a;
#pragma unroll
    for (int i = 0; i < 4; ++i) {
      a.p[i]     = __builtin_amdgcn_cvt_pkrtz(xlo[2 * i], xlo[2 * i + 1]);
      a.p[4 + i] = __builtin_amdgcn_cvt_pkrtz(xhi[2 * i], xhi[2 * i + 1]);
    }

#pragma unroll
    for (int nt = 0; nt < 4; ++nt) {
      const size_t wrow = (size_t)(nt * 16 + m) * CC + k0 + boff;
      v16h bq = *(const v16h*)(wqt + wrow);
      v16h bk = *(const v16h*)(wkt + wrow);
      v16h bv = *(const v16h*)(wvt + wrow);
      accq[nt] = __builtin_amdgcn_wmma_f32_16x16x32_f16(false, a.v, false, bq, (short)0, accq[nt], false, false);
      acck[nt] = __builtin_amdgcn_wmma_f32_16x16x32_f16(false, a.v, false, bk, (short)0, acck[nt], false, false);
      accv[nt] = __builtin_amdgcn_wmma_f32_16x16x32_f16(false, a.v, false, bv, (short)0, accv[nt], false, false);
    }
  }

  // Store results (C/D layout: VGPR r -> M = r + 8*hiL, lane -> N)
#pragma unroll
  for (int nt = 0; nt < 4; ++nt) {
#pragma unroll
    for (int r = 0; r < 8; ++r) {
      const int t = r0 + r + (hiL ? 8 : 0);
      const int n = nt * 16 + m;
      qh[(size_t)(b * TT + t) * HS + n] = (_Float16)accq[nt][r];
      kh[(size_t)(b * TT + t) * HS + n] = (_Float16)acck[nt][r];
      vt[(size_t)b * HS * TT + (size_t)n * TT + t] = (_Float16)accv[nt][r];
    }
  }
}

// ---------------------------------------------------------------------------
// Causal flash attention: one wave32 per 16-row tile of T per batch.
// Loops KV in 32-wide blocks up to the diagonal; online softmax in f32;
// P relaid C/D -> A layout through a 1KB LDS tile.
// ---------------------------------------------------------------------------
__global__ __launch_bounds__(32) void attn_wmma_kernel(
    const _Float16* __restrict__ qh,   // [B][T][HS]
    const _Float16* __restrict__ kh,   // [B][T][HS]
    const _Float16* __restrict__ vt,   // [B][HS][T]
    float* __restrict__ out)           // [B][T][HS] f32
{
  __shared__ _Float16 Pl[16][32];

  const int lane = threadIdx.x;
  const int tile = blockIdx.x % (TT / 16);
  const int b    = blockIdx.x / (TT / 16);
  const int ts0  = tile * 16;

  const int m    = lane & 15;
  const int hiL  = lane >> 4;
  const int aoff = hiL ? 16 : 0;
  const int boff = hiL ? 16 : 0;

  // Q as two 16x32 A-fragments (H = 64); union aliases the 16B loads directly
  AFrag qf[2];
  const _Float16* qrow = qh + (size_t)(b * TT + ts0 + m) * HS;
#pragma unroll
  for (int hc = 0; hc < 2; ++hc) {
    qf[hc].s.lo = *(const v8h*)(qrow + hc * 32 + aoff);
    qf[hc].s.hi = *(const v8h*)(qrow + hc * 32 + aoff + 8);
  }

  v8f o[4] = {};
  float mrow[8], lrow[8];
#pragma unroll
  for (int r = 0; r < 8; ++r) { mrow[r] = -1e30f; lrow[r] = 0.0f; }

  const float scale = 0.125f;  // 1/sqrt(64)

  for (int tk0 = 0; tk0 <= ts0 + 15; tk0 += 32) {
    // Prefetch next K block (L2-resident anyway; cheap hint)
    if (tk0 + 32 < TT)
      __builtin_prefetch(kh + (size_t)(b * TT + tk0 + 32 + m) * HS, 0, 0);

    // ---- S = Q K^T : two 16x16 subtiles, K-dim = H in chunks of 32 ----
    v8f s[2];
#pragma unroll
    for (int jj = 0; jj < 2; ++jj) {
      v8f acc = {};
#pragma unroll
      for (int hc = 0; hc < 2; ++hc) {
        const _Float16* kp = kh + (size_t)(b * TT + tk0 + jj * 16 + m) * HS + hc * 32 + boff;
        v16h bf = *(const v16h*)kp;
        acc = __builtin_amdgcn_wmma_f32_16x16x32_f16(false, qf[hc].v, false, bf, (short)0, acc, false, false);
      }
      s[jj] = acc;
    }

    // ---- scale + causal mask ----
#pragma unroll
    for (int jj = 0; jj < 2; ++jj) {
#pragma unroll
      for (int r = 0; r < 8; ++r) {
        const int ts = ts0 + r + (hiL ? 8 : 0);
        const int tk = tk0 + jj * 16 + m;
        const float sv = s[jj][r] * scale;
        s[jj][r] = (tk <= ts) ? sv : -1e30f;
      }
    }

    // ---- online softmax (rows live across 16 lanes of a half-group) ----
    float mnew[8], alpha[8];
#pragma unroll
    for (int r = 0; r < 8; ++r) {
      float v = fmaxf(s[0][r], s[1][r]);
      v = fmaxf(v, __shfl_xor(v, 1, 32));
      v = fmaxf(v, __shfl_xor(v, 2, 32));
      v = fmaxf(v, __shfl_xor(v, 4, 32));
      v = fmaxf(v, __shfl_xor(v, 8, 32));
      mnew[r]  = fmaxf(mrow[r], v);
      alpha[r] = __expf(mrow[r] - mnew[r]);
      mrow[r]  = mnew[r];
    }

#pragma unroll
    for (int r = 0; r < 8; ++r) {
      const float p0 = __expf(s[0][r] - mnew[r]);
      const float p1 = __expf(s[1][r] - mnew[r]);
      s[0][r] = p0; s[1][r] = p1;
      float rs = p0 + p1;
      rs += __shfl_xor(rs, 1, 32);
      rs += __shfl_xor(rs, 2, 32);
      rs += __shfl_xor(rs, 4, 32);
      rs += __shfl_xor(rs, 8, 32);
      lrow[r] = lrow[r] * alpha[r] + rs;
#pragma unroll
      for (int nt = 0; nt < 4; ++nt) o[nt][r] *= alpha[r];
    }

    // ---- relayout P: C/D layout -> A layout via LDS ----
    __syncthreads();   // WAR vs previous iteration's reads
#pragma unroll
    for (int r = 0; r < 8; ++r) {
      const int M = r + (hiL ? 8 : 0);
      Pl[M][m]      = (_Float16)s[0][r];
      Pl[M][m + 16] = (_Float16)s[1][r];
    }
    __syncthreads();

    AFrag pf;
#pragma unroll
    for (int i = 0; i < 8; ++i) {
      pf.v[i]     = Pl[m][aoff + i];
      pf.v[i + 8] = Pl[m][aoff + 8 + i];
    }

    // ---- O += P * V  (V fragment contiguous thanks to [H][T] layout) ----
#pragma unroll
    for (int nt = 0; nt < 4; ++nt) {
      const _Float16* vp = vt + (size_t)b * HS * TT + (size_t)(nt * 16 + m) * TT + tk0 + boff;
      v16h bf = *(const v16h*)vp;
      o[nt] = __builtin_amdgcn_wmma_f32_16x16x32_f16(false, pf.v, false, bf, (short)0, o[nt], false, false);
    }
  }

  // ---- normalize (v_rcp_f32 + mul instead of 32 divides) and store f32 ----
  float linv[8];
#pragma unroll
  for (int r = 0; r < 8; ++r) linv[r] = __builtin_amdgcn_rcpf(lrow[r]);

#pragma unroll
  for (int nt = 0; nt < 4; ++nt) {
#pragma unroll
    for (int r = 0; r < 8; ++r) {
      const int t = ts0 + r + (hiL ? 8 : 0);
      const int n = nt * 16 + m;
      out[(size_t)(b * TT + t) * HS + n] = o[nt][r] * linv[r];
    }
  }
}

// ---------------------------------------------------------------------------
extern "C" void kernel_launch(void* const* d_in, const int* in_sizes, int n_in,
                              void* d_out, int out_size, void* d_ws, size_t ws_size,
                              hipStream_t stream) {
  const float* x  = (const float*)d_in[0];
  const float* Wq = (const float*)d_in[1];
  const float* Wk = (const float*)d_in[2];
  const float* Wv = (const float*)d_in[3];
  float* out = (float*)d_out;

  // Workspace layout (f16): 3 transposed weights + q,k (row-major) + v (transposed)
  _Float16* ws  = (_Float16*)d_ws;
  _Float16* wqt = ws;                                    // HS*CC
  _Float16* wkt = wqt + (size_t)HS * CC;                 // HS*CC
  _Float16* wvt = wkt + (size_t)HS * CC;                 // HS*CC
  _Float16* qh  = wvt + (size_t)HS * CC;                 // BB*TT*HS
  _Float16* kh  = qh  + (size_t)BB * TT * HS;            // BB*TT*HS
  _Float16* vt  = kh  + (size_t)BB * TT * HS;            // BB*HS*TT

  cvt_w_kernel<<<(HS * CC + 255) / 256, 256, 0, stream>>>(Wq, Wk, Wv, wqt, wkt, wvt);

  qkv_wmma_kernel<<<BB * (TT / 16), 32, 0, stream>>>(x, wqt, wkt, wvt, qh, kh, vt);

  attn_wmma_kernel<<<BB * (TT / 16), 32, 0, stream>>>(qh, kh, vt, out);
}
